// ModulatedConv2d_23948737643023
// MI455X (gfx1250) — matrix operand order repaired and checked by reading, not compiled, not run on hardware
//
#include <hip/hip_runtime.h>

typedef __attribute__((ext_vector_type(16))) __bf16 v16bf;
typedef __attribute__((ext_vector_type(8)))  __bf16 v8bf;
typedef __attribute__((ext_vector_type(8)))  float  v8f;

#define IN_F  512
#define OUT_F 512
#define HH    64
#define WW    64
#define XROW  (WW + 2)                 /* 66: 1-col zero halo each side */
#define BATCH 16
#define EPSV  1e-8f
#define C_EQ  0.014731391274719738f    /* 1/sqrt(512*9) */

#define BCH    32                      /* ic chunk per stage */
#define LSTR   40                      /* LDS col stride in bf16: 20 words -> bank-optimal */
#define NUNIT  (XROW * (BCH / 8))      /* 264 16-byte units per stage */

// ---------------- prep 1: weight -> bf16 [tap][oc][ic], and wsq[oc][ic] ----------------
__global__ void prep_weight_kernel(const float* __restrict__ weight,
                                   __bf16* __restrict__ w16,
                                   float* __restrict__ wsq) {
  int tid = blockIdx.x * blockDim.x + threadIdx.x;          // oc*IN_F + ic
  int oc = tid / IN_F, ic = tid % IN_F;
  const float* wp = weight + (size_t)tid * 9;
  float ss = 0.f;
#pragma unroll
  for (int k = 0; k < 9; ++k) {
    float v = wp[k];
    ss += v * v;
    w16[((size_t)k * OUT_F + oc) * IN_F + ic] = (__bf16)v;
  }
  wsq[tid] = ss;
}

// ---------------- prep 2: xs[b][y][x+1][ic] = bf16( x[b][ic][y][x] * s[b][ic] ) --------
__global__ void prep_xs_kernel(const float* __restrict__ x,
                               const float* __restrict__ s,
                               __bf16* __restrict__ xs) {
  size_t tid = (size_t)blockIdx.x * blockDim.x + threadIdx.x;
  int w  = (int)(tid) & (WW - 1);
  int h  = (int)(tid >> 6) & (HH - 1);
  int ic = (int)(tid >> 12) & (IN_F - 1);
  int b  = (int)(tid >> 21);
  float v = x[tid] * s[b * IN_F + ic];
  xs[((size_t)(b * HH + h) * XROW + (w + 1)) * IN_F + ic] = (__bf16)v;
}

// ---------------- prep 2b: zero-fill the x = -1 and x = 64 halo columns ----------------
__global__ void halo_zero_kernel(__bf16* __restrict__ xs) {
  int tid = blockIdx.x * blockDim.x + threadIdx.x;   // BATCH*HH*2*IN_F = 2^20
  int ic   = tid & (IN_F - 1);
  int side = (tid >> 9) & 1;
  int h    = (tid >> 10) & (HH - 1);
  int b    = tid >> 16;
  xs[((size_t)(b * HH + h) * XROW + side * (XROW - 1)) * IN_F + ic] = (__bf16)0.0f;
}

// ---------------- prep 3: scale[b][oc] = C_EQ * rsqrt(C_EQ^2 * sum_ic s^2*wsq + eps) ---
__global__ void prep_scale_kernel(const float* __restrict__ s,
                                  const float* __restrict__ wsq,
                                  float* __restrict__ scale) {
  int tid = blockIdx.x * blockDim.x + threadIdx.x;           // b*OUT_F + oc
  int b = tid / OUT_F, oc = tid % OUT_F;
  const float* sp = s + b * IN_F;
  const float* wq = wsq + (size_t)oc * IN_F;
  float sum = 0.f;
  for (int ic = 0; ic < IN_F; ++ic) {
    float sv = sp[ic];
    sum += sv * sv * wq[ic];
  }
  scale[tid] = C_EQ * rsqrtf(C_EQ * C_EQ * sum + EPSV);
}

// ---------------- main: implicit-GEMM modulated conv, WMMA bf16 + async-LDS B ----------
// grid: (OUT_F/128, HH, BATCH), block 256 (8 waves).
// Wave w owns the unique oc tile m0 = m0blk + w*16 and all four 16-col n-tiles (1x4).
// Per stage (dy, ic-chunk): GLOBAL_LOAD_ASYNC_TO_LDS_B128 prefetches the haloed row chunk
// xs[b][y+dy][0..65][ic0..ic0+31] for stage s+1 (double-buffered, ASYNCcnt-tracked, no
// VGPR roundtrip) while the wave runs 3 dx * (1 A global frag + 4 B LDS frags + 4 WMMAs)
// on stage s. s_wait_asynccnt 0 + barrier publish the prefetch after compute.
__global__ __launch_bounds__(256) void modconv_wmma_kernel(
    const __bf16* __restrict__ w16,    // [9][OUT_F][IN_F]
    const __bf16* __restrict__ xs,     // [B][H][XROW][IN_F] with zero halo
    const float*  __restrict__ scale,  // [B][OUT_F]
    const float*  __restrict__ bias,   // [OUT_F]
    float* __restrict__ out)           // [B][OUT_F][H][W]
{
  __shared__ __bf16 bsh[2][XROW * LSTR];   // 2 x 5280 B

  const int b     = blockIdx.z;
  const int y     = blockIdx.y;
  const int m0blk = blockIdx.x * 128;

  const int tid  = threadIdx.x;
  const int wave = tid >> 5;
  const int lane = tid & 31;
  const int lm   = lane & 15;   // row/col within 16
  const int hw   = lane >> 4;   // half-wave: selects K-subgroup

  const int m0 = m0blk + wave * 16;   // this wave's unique oc tile

  v8f acc[4] = {{}, {}, {}, {}};      // n-tiles at cols 0,16,32,48

  // valid dy range (contiguous: {0,1} at y=0, {-1,0} at y=63, else {-1,0,1})
  const int dylo = (y == 0) ? 0 : -1;
  const int dyhi = (y == HH - 1) ? 0 : 1;
  const int nstages = (dyhi - dylo + 1) * (IN_F / BCH);

  // async global->LDS prefetch of stage s into bsh[s&1] (no VGPR roundtrip)
  auto async_load = [&](int s) {
    const int dy  = dylo + (s >> 4);
    const int ic0 = (s & 15) * BCH;
    const __bf16* src = xs + (size_t)(b * HH + y + dy) * XROW * IN_F + ic0;
    __bf16* dst = bsh[s & 1];
    {
      int col = tid >> 2, icq = tid & 3;
      unsigned int ldsa = (unsigned int)(uintptr_t)(dst + col * LSTR + icq * 8);
      const __bf16* g = src + (size_t)col * IN_F + icq * 8;
      asm volatile("global_load_async_to_lds_b128 %0, %1, off"
                   :: "v"(ldsa), "v"(g) : "memory");
    }
    if (tid < NUNIT - 256) {
      int u = tid + 256;
      int col = u >> 2, icq = u & 3;
      unsigned int ldsa = (unsigned int)(uintptr_t)(dst + col * LSTR + icq * 8);
      const __bf16* g = src + (size_t)col * IN_F + icq * 8;
      asm volatile("global_load_async_to_lds_b128 %0, %1, off"
                   :: "v"(ldsa), "v"(g) : "memory");
    }
  };

  async_load(0);
  asm volatile("s_wait_asynccnt 0x0" ::: "memory");
  __syncthreads();

  for (int s = 0; s < nstages; ++s) {
    if (s + 1 < nstages) async_load(s + 1);   // in flight during compute

    const int dy  = dylo + (s >> 4);
    const int ic0 = (s & 15) * BCH;
    const int tapbase = (dy + 1) * 3;
    const __bf16* bbuf = bsh[s & 1];
    // B lane layout used here (contiguous k, same order as A): elems 0..15 -> K=hw*16+0..15
    const __bf16* bp0 = bbuf + (size_t)lm * LSTR + hw * 16;

#pragma unroll
    for (int dx = 0; dx < 3; ++dx) {
      const int tap = tapbase + dx;
      // A lane layout used here (contiguous k, same order as B): elems 0..15 -> K=hw*16+0..15
      const __bf16* arow = w16 + ((size_t)tap * OUT_F + m0 + lm) * IN_F + ic0 + hw * 16;
      v16bf A;
      ((v8bf*)&A)[0] = *(const v8bf*)(arow);
      ((v8bf*)&A)[1] = *(const v8bf*)(arow + 8);

      const __bf16* bp = bp0 + dx * LSTR;   // col = lm + dx (halo already shifts by +1)
#pragma unroll
      for (int ni = 0; ni < 4; ++ni) {
        v16bf B;
        ((v8bf*)&B)[0] = *(const v8bf*)(bp + ni * 16 * LSTR);
        ((v8bf*)&B)[1] = *(const v8bf*)(bp + ni * 16 * LSTR + 8);
        acc[ni] = __builtin_amdgcn_wmma_f32_16x16x32_bf16(false, A, false, B,
                                                          (short)0, acc[ni], false, false);
      }
    }

    asm volatile("s_wait_asynccnt 0x0" ::: "memory");  // prefetch landed in LDS
    __syncthreads();                                    // publish to all waves
  }

  // epilogue: C/D layout -> lane holds N=lm, M = hw*8 + r (r = VGPR index)
#pragma unroll
  for (int r = 0; r < 8; ++r) {
    const int ocr = m0 + hw * 8 + r;
    const float sc = scale[b * OUT_F + ocr];
    const float bi = bias[ocr];
    float* o = out + (((size_t)b * OUT_F + ocr) * HH + y) * WW;
#pragma unroll
    for (int ni = 0; ni < 4; ++ni)
      o[ni * 16 + lm] = acc[ni][r] * sc + bi;
  }
}

extern "C" void kernel_launch(void* const* d_in, const int* in_sizes, int n_in,
                              void* d_out, int out_size, void* d_ws, size_t ws_size,
                              hipStream_t stream) {
  (void)in_sizes; (void)n_in; (void)out_size; (void)ws_size;
  const float* x      = (const float*)d_in[0];
  const float* s      = (const float*)d_in[1];
  const float* weight = (const float*)d_in[2];
  const float* bias   = (const float*)d_in[3];
  float* out = (float*)d_out;

  char* ws = (char*)d_ws;
  // 256-aligned workspace carve-up (~75 MB total)
  __bf16* w16   = (__bf16*)(ws);                                    // 9*512*512*2     = 4,718,592 B
  __bf16* xs    = (__bf16*)(ws + 4718592);                          // 16*64*66*512*2  = 69,206,016 B
  float*  wsq   = (float*) (ws + 4718592 + 69206016);               // 512*512*4       = 1,048,576 B
  float*  scale = (float*) (ws + 4718592 + 69206016 + 1048576);     // 16*512*4        = 32,768 B

  prep_weight_kernel<<<(OUT_F * IN_F) / 256, 256, 0, stream>>>(weight, w16, wsq);
  prep_xs_kernel<<<(BATCH * IN_F * HH * WW) / 256, 256, 0, stream>>>(x, s, xs);
  halo_zero_kernel<<<(BATCH * HH * 2 * IN_F) / 256, 256, 0, stream>>>(xs);
  prep_scale_kernel<<<(BATCH * OUT_F) / 256, 256, 0, stream>>>(s, wsq, scale);

  dim3 grid(OUT_F / 128, HH, BATCH);
  modconv_wmma_kernel<<<grid, 256, 0, stream>>>(w16, xs, scale, bias, out);
}
